// Generator_32427003085023
// MI455X (gfx1250) — matrix operand order repaired and checked by reading, not compiled, not run on hardware
//
#include <hip/hip_runtime.h>
#include <math.h>

typedef _Float16 half8 __attribute__((ext_vector_type(8)));
typedef _Float16 v16h  __attribute__((ext_vector_type(16)));
typedef float    v8f   __attribute__((ext_vector_type(8)));

#define ACT_NONE    0
#define ACT_LRELU   1
#define ACT_SIGMOID 2

static __device__ __forceinline__ float lrelu_f(float v) { return v >= 0.f ? v : 0.2f * v; }

// ---------------- weight convert + pad (f32 -> f16, [CoutPad][Kpad]) ----------------
__global__ void k_cvt_w(const float* __restrict__ w, _Float16* __restrict__ o,
                        int Cout, int K, int CoutPad, int Kpad) {
  int id = blockIdx.x * blockDim.x + threadIdx.x;
  int total = CoutPad * Kpad;
  if (id >= total) return;
  int co = id / Kpad, k = id - co * Kpad;
  float v = (co < Cout && k < K) ? w[co * K + k] : 0.f;
  o[id] = (_Float16)v;
}

// ---------------- zero fill ----------------
__global__ void k_zero(float* __restrict__ p, size_t n) {
  size_t id = (size_t)blockIdx.x * blockDim.x + threadIdx.x;
  if (id < n) p[id] = 0.f;
}

// ---------------- enc conv1: 1 -> 64 ch, 8x8, lrelu (direct) ----------------
__global__ void k_enc1(const float* __restrict__ in, const float* __restrict__ w,
                       const float* __restrict__ b, float* __restrict__ o, int Nn) {
  size_t id = (size_t)blockIdx.x * blockDim.x + threadIdx.x;
  size_t total = (size_t)Nn * 64 * 64;
  if (id >= total) return;
  int p  = (int)(id & 63);
  int co = (int)((id >> 6) & 63);
  size_t img = id >> 12;
  int y = p >> 3, x = p & 7;
  const float* ip = in + img * 64;
  float acc = b[co];
#pragma unroll
  for (int t = 0; t < 9; ++t) {
    int dy = t / 3 - 1, dx = t % 3 - 1;
    int yy = y + dy, xc = x + dx;
    if (yy >= 0 && yy < 8 && xc >= 0 && xc < 8)
      acc += ip[yy * 8 + xc] * w[co * 9 + t];
  }
  o[(img * 64 + (size_t)co) * 64 + p] = lrelu_f(acc);
}

// ---------------- build zy = [z[nd], given_y] as f16, K padded 147->160 ----------------
__global__ void k_zy(const float* __restrict__ z, const float* __restrict__ gy,
                     const int* __restrict__ nd, _Float16* __restrict__ zy, int Nn) {
  size_t id = (size_t)blockIdx.x * blockDim.x + threadIdx.x;
  size_t total = (size_t)Nn * 160;
  if (id >= total) return;
  int i = (int)(id / 160), k = (int)(id - (size_t)i * 160);
  float v = 0.f;
  if (k < 128)      v = z[(size_t)nd[i] * 128 + k];
  else if (k < 147) v = gy[(size_t)i * 19 + (k - 128)];
  zy[id] = (_Float16)v;
}

// ---------------- l1 GEMM: x[i][j] += zy[i] . l1_w[j] + bias[j] (WMMA) ----------------
__global__ __launch_bounds__(32) void k_l1_gemm(
    const _Float16* __restrict__ zy, const _Float16* __restrict__ wf,
    const float* __restrict__ bias, float* __restrict__ x, int Kpad) {
  const int lane = threadIdx.x;
  const int j0 = blockIdx.x * 16;
  const int i0 = blockIdx.y * 16;
  const _Float16* arow = zy + (size_t)(i0 + (lane & 15)) * Kpad;
  const _Float16* brow = wf + (size_t)(j0 + (lane & 15)) * Kpad;
  const int aoff = (lane & 16) ? 8 : 0;
  const int boff = (lane & 16) ? 8 : 0;
  v8f acc = {};
  for (int k0 = 0; k0 < Kpad; k0 += 32) {
    half8 a0 = *(const half8*)(arow + k0 + aoff);
    half8 a1 = *(const half8*)(arow + k0 + 16 + aoff);
    half8 b0 = *(const half8*)(brow + k0 + boff);
    half8 b1 = *(const half8*)(brow + k0 + 16 + boff);
    v16h a = __builtin_shufflevector(a0, a1, 0,1,2,3,4,5,6,7,8,9,10,11,12,13,14,15);
    v16h b = __builtin_shufflevector(b0, b1, 0,1,2,3,4,5,6,7,8,9,10,11,12,13,14,15);
    acc = __builtin_amdgcn_wmma_f32_16x16x32_f16(false, a, false, b, (short)0, acc, false, false);
  }
  const int n = lane & 15;
  const int mbase = (lane & 16) ? 8 : 0;
#pragma unroll
  for (int r = 0; r < 8; ++r) {
    int i = i0 + mbase + r, j = j0 + n;
    size_t idx = (size_t)i * 2048 + j;
    x[idx] = x[idx] + acc[r] + bias[j];
  }
}

// ---------------- generic 3x3 SAME conv via WMMA implicit GEMM ----------------
// One wave per block; tile = 16 out-channels x 16 pixels. Optional: 3-way channel-concat
// inputs, upsample-by-2 input read, bias, BN affine, activation, post-activation residual.
__global__ __launch_bounds__(32) void k_conv3x3_wmma(
    const float* __restrict__ in0, const float* __restrict__ in1, const float* __restrict__ in2,
    int cin_per, int Cin, int Cout, int H, int upsample,
    const _Float16* __restrict__ wf, int Kpad,
    const float* __restrict__ bias,
    const float* __restrict__ bnG, const float* __restrict__ bnB,
    const float* __restrict__ res,
    float* __restrict__ out, int act) {
  extern __shared__ _Float16 patch[];   // [Cin][PR][PC] f16, zero-padded borders
  const int lane = threadIdx.x;
  const int W = H;
  const int tileW = (H < 16) ? H : 16;  // pixels per tile row
  const int PR = (16 / tileW) + 2;      // patch rows
  const int PC = tileW + 2;             // patch cols
  const int img = blockIdx.z;
  const int co0 = blockIdx.y * 16;
  const int n0  = blockIdx.x * 16;
  const int rowStart = n0 / W;
  const int colStart = n0 - rowStart * W;
  const int Hin = upsample ? (H >> 1) : H;
  const int inCh = Hin * Hin;

  // --- stage input patch to LDS (f32 -> f16) ---
  const int patchElems = Cin * PR * PC;
  for (int idx = lane; idx < patchElems; idx += 32) {
    int ci  = idx / (PR * PC);
    int rem = idx - ci * (PR * PC);
    int r = rem / PC;
    int c = rem - r * PC;
    int gy = rowStart - 1 + r;
    int gx = colStart - 1 + c;
    float v = 0.f;
    if (gy >= 0 && gy < H && gx >= 0 && gx < W) {
      int iy = upsample ? (gy >> 1) : gy;
      int ix = upsample ? (gx >> 1) : gx;
      const float* src = in0; int cc = ci;
      if (ci >= cin_per) {
        if (ci < 2 * cin_per) { src = in1; cc = ci - cin_per; }
        else                  { src = in2; cc = ci - 2 * cin_per; }
      }
      v = src[((size_t)img * cin_per + cc) * inCh + (size_t)iy * Hin + ix];
    }
    patch[idx] = (_Float16)v;
  }
  __syncthreads();

  // --- K loop, 32 per WMMA step ---
  const _Float16* wrow = wf + (size_t)(co0 + (lane & 15)) * Kpad;
  const int aoff = (lane & 16) ? 16 : 0;  // A: lanes>=16 hold K 16..31
  const int boff = (lane & 16) ? 16 : 0;  // B: lanes>=16 hold K 16..31
  const int n  = lane & 15;
  const int lr = n / tileW;
  const int lc = n - lr * tileW;
  v8f acc = {};
  for (int k0 = 0; k0 < Kpad; k0 += 32) {
    half8 a0 = *(const half8*)(wrow + k0 + aoff);
    half8 a1 = *(const half8*)(wrow + k0 + aoff + 8);
    v16h a = __builtin_shufflevector(a0, a1, 0,1,2,3,4,5,6,7,8,9,10,11,12,13,14,15);
    v16h b;
#pragma unroll
    for (int i = 0; i < 16; ++i) {
      int k  = k0 + boff + i;      // k = ci*9 + t
      int ci = k / 9;
      int t  = k - ci * 9;
      int dy = t / 3;              // 0..2 -> patch row lr+dy (border already included)
      int dx = t - dy * 3;         // 0..2
      b[i] = patch[ci * (PR * PC) + (lr + dy) * PC + (lc + dx)];
    }
    acc = __builtin_amdgcn_wmma_f32_16x16x32_f16(false, a, false, b, (short)0, acc, false, false);
  }

  // --- epilogue: bias, BN, activation, THEN residual (x = x + lrelu(conv)), store ---
  const int mbase = (lane & 16) ? 8 : 0;
  const int py = rowStart + lr;
  const int px = colStart + lc;
#pragma unroll
  for (int r = 0; r < 8; ++r) {
    int co = co0 + mbase + r;
    if (co < Cout) {
      float v = acc[r] + bias[co];
      if (bnG) {
        float inv = bnG[co] * rsqrtf(1.f + 1e-5f);
        v = v * inv + bnB[co];
      }
      if (act == ACT_LRELU)        v = lrelu_f(v);
      else if (act == ACT_SIGMOID) v = 1.f / (1.f + __expf(-v));
      size_t oidx = ((size_t)img * Cout + co) * (size_t)(H * W) + (size_t)py * W + px;
      if (res) v += res[oidx];   // residual applied AFTER activation (matches reference)
      out[oidx] = v;
    }
  }
}

// ---------------- edge scatter: pp/pn += x[src] by edge sign ----------------
__global__ void k_scatter(const float* __restrict__ x, const int* __restrict__ gw,
                          float* __restrict__ pp, float* __restrict__ pn, int Nn) {
  int e = blockIdx.x;
  int typ = gw[e * 3 + 1];
  if (typ == 0) return;
  int src = gw[e * 3 + 0]; src = src < 0 ? 0 : (src > Nn - 1 ? Nn - 1 : src);
  int tgt = gw[e * 3 + 2]; tgt = tgt < 0 ? 0 : (tgt > Nn - 1 ? Nn - 1 : tgt);
  float* dst = (typ > 0) ? pp : pn;
  const float* s = x + (size_t)src * 2048;
  float* d = dst + (size_t)tgt * 2048;
  for (int j = threadIdx.x; j < 2048; j += blockDim.x)
    atomicAdd(&d[j], s[j]);
}

// ---------------- pool + self-attention per sample (B blocks x 32 threads) ----------------
__global__ __launch_bounds__(32) void k_attn(
    const float* __restrict__ x,
    const float* __restrict__ wq, const float* __restrict__ bq,
    const float* __restrict__ wk, const float* __restrict__ bk,
    const float* __restrict__ wv, const float* __restrict__ bv,
    float* __restrict__ agg) {
  __shared__ float xp[32][32], Q[32][32], Km[32][32], V[32][32], RA[32][32];
  int b = blockIdx.x, r = threadIdx.x;
  for (int c = 0; c < 32; ++c) {
    const float* p = x + (((size_t)(b * 32 + r)) * 32 + c) * 64;
    float s = 0.f;
    for (int q = 0; q < 64; ++q) s += p[q];
    xp[r][c] = s * (1.f / 64.f);
  }
  __syncthreads();
  for (int c = 0; c < 32; ++c) {
    float q = bq[c], kk = bk[c], v = bv[c];
    for (int d = 0; d < 32; ++d) {
      float xv = xp[r][d];
      q  += xv * wq[c * 32 + d];
      kk += xv * wk[c * 32 + d];
      v  += xv * wv[c * 32 + d];
    }
    Q[r][c] = q; Km[r][c] = kk; V[r][c] = v;
  }
  __syncthreads();
  float sc[32];
  float mx = -1e30f;
  const float scale = 1.f / sqrtf(32.f);
  for (int s = 0; s < 32; ++s) {
    float d = 0.f;
    for (int c = 0; c < 32; ++c) d += Q[r][c] * Km[s][c];
    d *= scale;
    sc[s] = d;
    mx = d > mx ? d : mx;
  }
  float se = 0.f;
  for (int s = 0; s < 32; ++s) { sc[s] = __expf(sc[s] - mx); se += sc[s]; }
  float inv = 1.f / se;
  for (int c = 0; c < 32; ++c) {
    float a = 0.f;
    for (int s = 0; s < 32; ++s) a += sc[s] * V[s][c];
    RA[r][c] = a * inv;
  }
  __syncthreads();
  float a = 0.f;
  for (int rr = 0; rr < 32; ++rr) a += RA[rr][threadIdx.x];
  agg[b * 32 + threadIdx.x] = a * (1.f / 32.f);
}

// ---------------- broadcast agg -> px (B,32,8,8) ----------------
__global__ void k_bcast(const float* __restrict__ agg, float* __restrict__ px, size_t total) {
  size_t id = (size_t)blockIdx.x * blockDim.x + threadIdx.x;
  if (id < total) px[id] = agg[id >> 6];
}

extern "C" void kernel_launch(void* const* d_in, const int* in_sizes, int n_in,
                              void* d_out, int out_size, void* d_ws, size_t ws_size,
                              hipStream_t stream) {
  (void)in_sizes; (void)n_in; (void)out_size; (void)ws_size;
  const float* z    = (const float*)d_in[0];
  const float* gy   = (const float*)d_in[1];
  const float* gm   = (const float*)d_in[2];
  const int*   gw   = (const int*)  d_in[3];
  const int*   nd   = (const int*)  d_in[4];
  const float* l1w  = (const float*)d_in[5];
  const float* l1b  = (const float*)d_in[6];
  const float* e1w1 = (const float*)d_in[7];
  const float* e1b1 = (const float*)d_in[8];
  const float* e1w2 = (const float*)d_in[9];
  const float* e1b2 = (const float*)d_in[10];
  const float* cw1  = (const float*)d_in[11];
  const float* cb1  = (const float*)d_in[12];
  const float* cw2  = (const float*)d_in[13];
  const float* cb2  = (const float*)d_in[14];
  const float* wq   = (const float*)d_in[15];
  const float* bq   = (const float*)d_in[16];
  const float* wk   = (const float*)d_in[17];
  const float* bk   = (const float*)d_in[18];
  const float* wv   = (const float*)d_in[19];
  const float* bv   = (const float*)d_in[20];
  const float* u1w  = (const float*)d_in[21];
  const float* u1b  = (const float*)d_in[22];
  const float* bn1g = (const float*)d_in[23];
  const float* bn1b = (const float*)d_in[24];
  const float* u2w  = (const float*)d_in[25];
  const float* u2b  = (const float*)d_in[26];
  const float* bn2g = (const float*)d_in[27];
  const float* bn2b = (const float*)d_in[28];
  const float* d3w  = (const float*)d_in[29];
  const float* d3b  = (const float*)d_in[30];
  const float* d4w  = (const float*)d_in[31];
  const float* d4b  = (const float*)d_in[32];
  const float* d5w  = (const float*)d_in[33];
  const float* d5b  = (const float*)d_in[34];
  float* out = (float*)d_out;

  const int Nn = 8192, Bs = 256, E = 65536;

  // -------- workspace carving --------
  char* ws = (char*)d_ws;
  size_t off = 0;
  auto alloc = [&](size_t bytes) -> char* {
    char* p = ws + off;
    off += (bytes + 255) & ~(size_t)255;
    return p;
  };
  float* x    = (float*)alloc((size_t)Nn * 2048 * 4);     // node features (m, then x)
  float* bufA = (float*)alloc((size_t)Nn * 64 * 64 * 4);  // enc1 out / h / up2 out
  float* pp   = (float*)alloc((size_t)Nn * 2048 * 4);     // pos msgs / d3 out
  float* pn   = (float*)alloc((size_t)Nn * 2048 * 4);     // neg msgs / d4 out
  float* px   = (float*)alloc((size_t)Bs * 32 * 64 * 4);
  float* y1   = (float*)alloc((size_t)Bs * 64 * 256 * 4);
  float* agg  = (float*)alloc((size_t)Bs * 32 * 4);
  _Float16* zy16 = (_Float16*)alloc((size_t)Nn * 160 * 2);
  _Float16* e2f  = (_Float16*)alloc((size_t)32 * 576 * 2);
  _Float16* c1f  = (_Float16*)alloc((size_t)3 * 64 * 864 * 2);
  _Float16* c2f  = (_Float16*)alloc((size_t)3 * 32 * 576 * 2);
  _Float16* u1f  = (_Float16*)alloc((size_t)64 * 288 * 2);
  _Float16* u2f  = (_Float16*)alloc((size_t)128 * 576 * 2);
  _Float16* d3f  = (_Float16*)alloc((size_t)64 * 1152 * 2);
  _Float16* d4f  = (_Float16*)alloc((size_t)32 * 576 * 2);
  _Float16* d5f  = (_Float16*)alloc((size_t)16 * 288 * 2);
  _Float16* l1f  = (_Float16*)alloc((size_t)2048 * 160 * 2);

  auto cvt = [&](const float* w, _Float16* o, int Cout, int K, int CoutPad, int Kpad) {
    int total = CoutPad * Kpad;
    k_cvt_w<<<(total + 255) / 256, 256, 0, stream>>>(w, o, Cout, K, CoutPad, Kpad);
  };
  cvt(e1w2, e2f, 32, 576, 32, 576);
  for (int i = 0; i < 3; ++i) {
    cvt(cw1 + (size_t)i * 64 * 864, c1f + (size_t)i * 64 * 864, 64, 864, 64, 864);
    cvt(cw2 + (size_t)i * 32 * 576, c2f + (size_t)i * 32 * 576, 32, 576, 32, 576);
  }
  cvt(u1w, u1f, 64, 288, 64, 288);
  cvt(u2w, u2f, 128, 576, 128, 576);
  cvt(d3w, d3f, 64, 1152, 64, 1152);
  cvt(d4w, d4f, 32, 576, 32, 576);
  cvt(d5w, d5f, 1, 288, 16, 288);
  cvt(l1w, l1f, 2048, 147, 2048, 160);

  auto conv = [&](const float* in0, const float* in1, const float* in2, int cin_per,
                  int Cin, int Cout, int H, int ups, const _Float16* wf, int Kp,
                  const float* bias, const float* bg, const float* bb,
                  const float* res, float* o, int act, int nImg) {
    int tileW = H < 16 ? H : 16;
    int PR = (16 / tileW) + 2, PC = tileW + 2;
    dim3 grid((H * H) / 16, (Cout + 15) / 16, nImg);
    size_t shm = (size_t)Cin * PR * PC * sizeof(_Float16);
    k_conv3x3_wmma<<<grid, 32, shm, stream>>>(in0, in1, in2, cin_per, Cin, Cout, H, ups,
                                              wf, Kp, bias, bg, bb, res, o, act);
  };

  // ---- encoder ----
  {
    size_t total = (size_t)Nn * 64 * 64;
    k_enc1<<<(unsigned)((total + 255) / 256), 256, 0, stream>>>(gm, e1w1, e1b1, bufA, Nn);
  }
  conv(bufA, nullptr, nullptr, 64, 64, 32, 8, 0, e2f, 576, e1b2,
       nullptr, nullptr, nullptr, x, ACT_LRELU, Nn);  // x = m

  // ---- x = m + (zy @ l1_w^T + b) ----
  {
    size_t total = (size_t)Nn * 160;
    k_zy<<<(unsigned)((total + 255) / 256), 256, 0, stream>>>(z, gy, nd, zy16, Nn);
    dim3 g(2048 / 16, Nn / 16);
    k_l1_gemm<<<g, 32, 0, stream>>>(zy16, l1f, l1b, x, 160);
  }

  // ---- message passing x3 ----
  for (int it = 0; it < 3; ++it) {
    size_t nelem = (size_t)Nn * 2048;
    k_zero<<<(unsigned)((nelem + 255) / 256), 256, 0, stream>>>(pp, nelem);
    k_zero<<<(unsigned)((nelem + 255) / 256), 256, 0, stream>>>(pn, nelem);
    k_scatter<<<E, 256, 0, stream>>>(x, gw, pp, pn, Nn);
    conv(x, pp, pn, 32, 96, 64, 8, 0, c1f + (size_t)it * 64 * 864, 864, cb1 + it * 64,
         nullptr, nullptr, nullptr, bufA, ACT_LRELU, Nn);                 // h = lrelu(conv1)
    conv(bufA, nullptr, nullptr, 64, 64, 32, 8, 0, c2f + (size_t)it * 32 * 576, 576,
         cb2 + it * 32, nullptr, nullptr, x, x, ACT_LRELU, Nn);           // x = x + lrelu(conv2(h))
  }

  // ---- pool + attention ----
  k_attn<<<Bs, 32, 0, stream>>>(x, wq, bq, wk, bk, wv, bv, agg);
  {
    size_t total = (size_t)Bs * 2048;
    k_bcast<<<(unsigned)((total + 255) / 256), 256, 0, stream>>>(agg, px, total);
  }

  // ---- decoder ----
  conv(px,   nullptr, nullptr,  32,  32,  64, 16, 1, u1f,  288, u1b, bn1g, bn1b, nullptr, y1,   ACT_LRELU,   Bs);
  conv(y1,   nullptr, nullptr,  64,  64, 128, 32, 1, u2f,  576, u2b, bn2g, bn2b, nullptr, bufA, ACT_LRELU,   Bs);
  conv(bufA, nullptr, nullptr, 128, 128,  64, 32, 0, d3f, 1152, d3b, nullptr, nullptr, nullptr, pp, ACT_LRELU, Bs);
  conv(pp,   nullptr, nullptr,  64,  64,  32, 32, 0, d4f,  576, d4b, nullptr, nullptr, nullptr, pn, ACT_LRELU, Bs);
  conv(pn,   nullptr, nullptr,  32,  32,   1, 32, 0, d5f,  288, d5b, nullptr, nullptr, nullptr, out, ACT_SIGMOID, Bs);
}
